// NPCFace_5205500363386
// MI455X (gfx1250) — matrix operand order repaired and checked by reading, not compiled, not run on hardware
//
#include <hip/hip_runtime.h>
#include <hip/hip_bf16.h>
#include <math.h>

typedef __attribute__((ext_vector_type(16))) _Float16 v16h;
typedef __attribute__((ext_vector_type(8)))  float    v8f;

#define S_SCALE 32.0f
#define T_BOOST 1.1f
#define ALPHA_B 0.25f
#define M0_C    0.4f
#define M1_C    0.2f
// cos(0.4), sin(0.4)
#define COS_M0  0.9210609940028851f
#define SIN_M0  0.3894183423086505f
// fixed logsumexp max: S * (T*1 + ALPHA) = 32 * 1.35
#define MAXL    43.2f

#define FDIM  512
#define BATCH 512
#define CLS   100000
#define NTILES (CLS / 16)      // 6250
#define CHUNK_T 8              // 8 N-tiles (128 classes) per block

__device__ __forceinline__ float waveSum32(float v) {
#pragma unroll
  for (int off = 16; off >= 1; off >>= 1) v += __shfl_xor(v, off, 32);
  return v;
}

// ---------------------------------------------------------------------------
// Kernel 1: normalize x rows, write xn (f32) and A-fragment-packed f16,
// and zero the 3*BATCH global accumulators.
// grid = 64 blocks x 256 threads; one wave per row.
// ---------------------------------------------------------------------------
__global__ void __launch_bounds__(256) npc_xprep(
    const float* __restrict__ x, _Float16* __restrict__ Apack,
    float* __restrict__ xn, float* __restrict__ accZero) {
  int gtid = blockIdx.x * blockDim.x + threadIdx.x;
  if (gtid < 3 * BATCH) accZero[gtid] = 0.0f;

  int wave = threadIdx.x >> 5, lane = threadIdx.x & 31;
  int r = blockIdx.x * 8 + wave;                 // 0..511
  const float* xr = x + (size_t)r * FDIM + lane * 16;
  float v[16];
  float ss = 0.0f;
#pragma unroll
  for (int i = 0; i < 16; ++i) { v[i] = xr[i]; ss += v[i] * v[i]; }
  ss = waveSum32(ss);
  float inv = 1.0f / fmaxf(sqrtf(ss), 1e-12f);

  int mtile = r >> 4, mrow = r & 15;
  int kbase = lane * 16;
#pragma unroll
  for (int i = 0; i < 16; ++i) {
    float xv = v[i] * inv;
    int k = kbase + i;
    xn[(size_t)r * FDIM + k] = xv;
    // A-fragment layout (16-bit A 16x32): lane L = mrow + 16*g,
    // half h: K = (h<8 ? h : h+8) + 8*g  (per ISA table)
    int ktile = k >> 5, kp = k & 31;
    int g = (kp >> 3) & 1;
    int h = (kp & 7) + ((kp >= 16) ? 8 : 0);
    int L = mrow + 16 * g;
    Apack[((size_t)(mtile * 16 + ktile) * 32 + L) * 16 + h] = (_Float16)xv;
  }
}

// ---------------------------------------------------------------------------
// Kernel 2: reciprocal row norms of W. grid = 12500 x 256; one wave per row.
// ---------------------------------------------------------------------------
__global__ void __launch_bounds__(256) npc_wprep(
    const float* __restrict__ W, float* __restrict__ rnorm) {
  int wave = threadIdx.x >> 5, lane = threadIdx.x & 31;
  int c = blockIdx.x * 8 + wave;                 // 0..99999
  const float* wr = W + (size_t)c * FDIM + lane * 16;
  float ss = 0.0f;
#pragma unroll
  for (int i = 0; i < 16; ++i) { float t = wr[i]; ss += t * t; }
  ss = waveSum32(ss);
  if (lane == 0) rnorm[c] = 1.0f / fmaxf(sqrtf(ss), 1e-12f);
}

// ---------------------------------------------------------------------------
// Kernel 3: gt = clip(<xn_r, Wn_label>), thresh = cos(acos(gt)+M0).
// grid = 64 x 256; one wave per row.
// ---------------------------------------------------------------------------
__global__ void __launch_bounds__(256) npc_gtprep(
    const float* __restrict__ xn, const float* __restrict__ W,
    const float* __restrict__ rnorm, const int* __restrict__ label,
    float* __restrict__ gtv, float* __restrict__ thv) {
  int wave = threadIdx.x >> 5, lane = threadIdx.x & 31;
  int r = blockIdx.x * 8 + wave;
  int c = label[r];
  const float* xr = xn + (size_t)r * FDIM + lane * 16;
  const float* wr = W + (size_t)c * FDIM + lane * 16;
  float d = 0.0f;
#pragma unroll
  for (int i = 0; i < 16; ++i) d += xr[i] * wr[i];
  d = waveSum32(d);
  if (lane == 0) {
    float g = fminf(fmaxf(d * rnorm[c], -1.0f), 1.0f);
    gtv[r] = g;
    float s = sqrtf(fmaxf(0.0f, 1.0f - g * g));
    thv[r] = g * COS_M0 - s * SIN_M0;
  }
}

// ---------------------------------------------------------------------------
// Kernel 4: fused WMMA GEMM + masking/boosting + fixed-max sum-exp.
// grid = (782, 4) x 256 threads (8 waves). Wave w owns 16-row M strip
// mtile = blockIdx.y*8 + w; block covers 8 N-tiles (128 classes).
// ---------------------------------------------------------------------------
__global__ void __launch_bounds__(256) npc_gemm(
    const float* __restrict__ W, const _Float16* __restrict__ Apack,
    const float* __restrict__ rnorm, const float* __restrict__ thv,
    const int* __restrict__ label,
    float* __restrict__ accE, float* __restrict__ accL,
    float* __restrict__ accC) {
  int wave = threadIdx.x >> 5, lane = threadIdx.x & 31;
  int mtile = blockIdx.y * 8 + wave;             // 0..31
  int grp = lane >> 4;                           // 0/1: C rows m = j + 8*grp
  int rbase = mtile * 16 + 8 * grp;

  int lb[8]; float th[8];
#pragma unroll
  for (int j = 0; j < 8; ++j) { lb[j] = label[rbase + j]; th[j] = thv[rbase + j]; }

  float aE[8], aL[8], aC[8];
#pragma unroll
  for (int j = 0; j < 8; ++j) { aE[j] = 0.0f; aL[j] = 0.0f; aC[j] = 0.0f; }

  const _Float16* abase = Apack + ((size_t)(mtile * 16) * 32 + lane) * 16;

  int nt0 = blockIdx.x * CHUNK_T;
  for (int t = 0; t < CHUNK_T; ++t) {
    int ntile = nt0 + t;
    if (ntile >= NTILES) break;
    int c = ntile * 16 + (lane & 15);            // B fragment: N = lane&15
    float rn = rnorm[c];
    const float* wrow = W + (size_t)c * FDIM + 8 * grp;   // K = 8*grp + (h<8 ? h : h+8)

    v8f acc = {0.f, 0.f, 0.f, 0.f, 0.f, 0.f, 0.f, 0.f};
#pragma unroll
    for (int kt = 0; kt < 16; ++kt) {
      // A fragment: packed, 32B per lane -> two b128 loads
      v16h a = *(const v16h*)(abase + (size_t)kt * 512);
      // B fragment built on the fly from fp32 W (normalized -> f16)
      const float* wp = wrow + kt * 32;
      float4 w0 = *(const float4*)(wp + 0);
      float4 w1 = *(const float4*)(wp + 4);
      float4 w2 = *(const float4*)(wp + 16);
      float4 w3 = *(const float4*)(wp + 20);
      v16h b;
      b[0]  = (_Float16)(w0.x * rn); b[1]  = (_Float16)(w0.y * rn);
      b[2]  = (_Float16)(w0.z * rn); b[3]  = (_Float16)(w0.w * rn);
      b[4]  = (_Float16)(w1.x * rn); b[5]  = (_Float16)(w1.y * rn);
      b[6]  = (_Float16)(w1.z * rn); b[7]  = (_Float16)(w1.w * rn);
      b[8]  = (_Float16)(w2.x * rn); b[9]  = (_Float16)(w2.y * rn);
      b[10] = (_Float16)(w2.z * rn); b[11] = (_Float16)(w2.w * rn);
      b[12] = (_Float16)(w3.x * rn); b[13] = (_Float16)(w3.y * rn);
      b[14] = (_Float16)(w3.z * rn); b[15] = (_Float16)(w3.w * rn);
      acc = __builtin_amdgcn_wmma_f32_16x16x32_f16(
          false, a, false, b, (short)0, acc, false, false);
    }

    // Epilogue on the 16x16 cos tile: clip, mask vs thresh, boost, sum-exp.
#pragma unroll
    for (int j = 0; j < 8; ++j) {
      float cv = fminf(fmaxf(acc[j], -1.0f), 1.0f);
      bool isgt = (c == lb[j]);
      bool hard = (cv > th[j]) && !isgt;
      float boosted = hard ? (T_BOOST * cv + ALPHA_B) : cv;
      if (!isgt) aE[j] += __expf(S_SCALE * boosted - MAXL);
      if (hard) { aC[j] += 1.0f; aL[j] += boosted; }
    }
  }

  // Reduce over the 16 lanes that share each row (same lane group).
#pragma unroll
  for (int j = 0; j < 8; ++j) {
#pragma unroll
    for (int off = 8; off >= 1; off >>= 1) {
      aE[j] += __shfl_xor(aE[j], off, 32);
      aL[j] += __shfl_xor(aL[j], off, 32);
      aC[j] += __shfl_xor(aC[j], off, 32);
    }
  }
  if ((lane & 15) == 0) {
#pragma unroll
    for (int j = 0; j < 8; ++j) {
      atomicAdd(&accE[rbase + j], aE[j]);
      atomicAdd(&accL[rbase + j], aL[j]);
      atomicAdd(&accC[rbase + j], aC[j]);
    }
  }
}

// ---------------------------------------------------------------------------
// Kernel 5: per-row margin + logsumexp close-out, focal CE. 1 block x 512.
// ---------------------------------------------------------------------------
__global__ void __launch_bounds__(512) npc_final(
    const float* __restrict__ accE, const float* __restrict__ accL,
    const float* __restrict__ accC, const float* __restrict__ gtv,
    float* __restrict__ out) {
  __shared__ float sdata[512];
  int r = threadIdx.x;
  float cnt = accC[r];
  float lam = accL[r] / fmaxf(cnt, 1.0f);
  float mi = M0_C + ((cnt > 0.0f) ? lam * M1_C : 0.0f);
  float g = gtv[r];
  float sm = sinf(mi), cm = cosf(mi);
  float cosm = g * cm - sqrtf(fmaxf(0.0f, 1.0f - g * g)) * sm;
  float fgt = (g > 0.0f) ? cosm : g;
  float gl = S_SCALE * fgt;
  float lse = MAXL + logf(accE[r] + __expf(gl - MAXL));
  sdata[r] = lse - gl;
  __syncthreads();
#pragma unroll
  for (int s = 256; s > 0; s >>= 1) {
    if (r < s) sdata[r] += sdata[r + s];
    __syncthreads();
  }
  if (r == 0) {
    float logp = sdata[0] / (float)BATCH;
    float p = __expf(-logp);
    float d = 1.0f - p;
    out[0] = d * d * logp;
  }
}

// ---------------------------------------------------------------------------
extern "C" void kernel_launch(void* const* d_in, const int* in_sizes, int n_in,
                              void* d_out, int out_size, void* d_ws, size_t ws_size,
                              hipStream_t stream) {
  (void)in_sizes; (void)n_in; (void)out_size; (void)ws_size;
  const float* x     = (const float*)d_in[0];
  const int*   label = (const int*)d_in[1];
  const float* W     = (const float*)d_in[2];

  char* ws = (char*)d_ws;
  _Float16* Apack = (_Float16*)ws;                         // 512 KB
  float* xn    = (float*)(ws + 524288);                    // 1 MB
  float* rnorm = (float*)(ws + 524288 + 1048576);          // 400 KB
  float* gtv   = rnorm + CLS;
  float* thv   = gtv + BATCH;
  float* accE  = thv + BATCH;                              // 3*512 contiguous
  float* accL  = accE + BATCH;
  float* accC  = accL + BATCH;

  npc_xprep<<<64, 256, 0, stream>>>(x, Apack, xn, accE);
  npc_wprep<<<CLS / 8, 256, 0, stream>>>(W, rnorm);
  npc_gtprep<<<64, 256, 0, stream>>>(xn, W, rnorm, label, gtv, thv);

  dim3 grid((NTILES + CHUNK_T - 1) / CHUNK_T, 4);
  npc_gemm<<<grid, 256, 0, stream>>>(W, Apack, rnorm, thv, label,
                                     accE, accL, accC);
  npc_final<<<1, 512, 0, stream>>>(accE, accL, accC, gtv, (float*)d_out);
}
